// NT_Xent_20255065768459
// MI455X (gfx1250) — compile-verified, matrix operand order checked by reading, not run on hardware
//
#include <hip/hip_runtime.h>
#include <hip/hip_bf16.h>

#define NROWS  8192
#define NDIM   256
#define MTILE  128         // rows per workgroup (8 waves x 16)
#define NTILE  64          // columns staged in LDS per iteration
#define CSPLIT 4           // column-dimension split for device occupancy
#define CTILES (NROWS / NTILE)
#define INV_T  2.0f        // 1 / temperature

typedef __attribute__((ext_vector_type(16))) __bf16 v16bf;
typedef __attribute__((ext_vector_type(8)))  float  v8f;
typedef __attribute__((ext_vector_type(4)))  unsigned int v4u;
typedef __attribute__((ext_vector_type(8)))  int v8i;
typedef __attribute__((ext_vector_type(4)))  int v4i;

#if defined(__has_builtin)
#if __has_builtin(__builtin_amdgcn_tensor_load_to_lds)
#define USE_TDM 1
#endif
#endif
#ifndef USE_TDM
#define USE_TDM 0
#endif

__device__ __forceinline__ unsigned short f2bf(float f) {
    union { float f; unsigned int u; } v; v.f = f;
    unsigned int u = v.u;
    unsigned int r = u + 0x7FFFu + ((u >> 16) & 1u);   // round to nearest even
    return (unsigned short)(r >> 16);
}

// ---------------------------------------------------------------------------
// Kernel 1: row-normalize z (fp32) -> zn (bf16), one wave per row.
// ---------------------------------------------------------------------------
__global__ __launch_bounds__(256) void ntxent_normalize(const float* __restrict__ z,
                                                        unsigned short* __restrict__ zn) {
    const int tid  = threadIdx.x;
    const int lane = tid & 31;
    const int wave = tid >> 5;
    const int row  = blockIdx.x * 8 + wave;

    const float* src = z + (size_t)row * NDIM;
    float vals[8];
    float ss = 0.f;
#pragma unroll
    for (int j = 0; j < 8; ++j) {
        float x = src[j * 32 + lane];
        vals[j] = x;
        ss += x * x;
    }
#pragma unroll
    for (int m = 16; m >= 1; m >>= 1) ss += __shfl_xor(ss, m, 32);

    const float inv = 1.0f / fmaxf(sqrtf(ss), 1e-8f);
    unsigned short* dst = zn + (size_t)row * NDIM;
#pragma unroll
    for (int j = 0; j < 8; ++j) dst[j * 32 + lane] = f2bf(vals[j] * inv);
}

#if USE_TDM
// Issue one TDM descriptor: DMA a 64-row x 256-col bf16 tile of zn into LDS
// double-buffer slot `buf`. 2D tensor, data_size=2B, global_addr = tile start.
__device__ __forceinline__ void tdm_stage(const unsigned short* zn, int ct, int buf) {
    const unsigned long long ga =
        (unsigned long long)(const void*)(zn + (size_t)ct * NTILE * NDIM);

    v4u g0;
    g0.x = 1u;                                         // count=1: valid user D#
    g0.y = (unsigned int)(buf * (NTILE * NDIM * 2));   // lds_addr (bytes)
    g0.z = (unsigned int)(ga & 0xFFFFFFFFu);           // global_addr[31:0]
    g0.w = (unsigned int)((ga >> 32) & 0x01FFFFFFu) | (2u << 30);  // addr[56:32] | type=2

    v8i g1;
    g1[0] = (int)(1u << 16);                           // data_size = 2 bytes
    g1[1] = (int)(((unsigned)NDIM & 0xFFFFu) << 16);   // tensor_dim0[15:0]
    g1[2] = (int)((((unsigned)NDIM >> 16) & 0xFFFFu) |
                  (((unsigned)NROWS & 0xFFFFu) << 16));// dim0[31:16] | dim1[15:0]
    g1[3] = (int)((((unsigned)NROWS >> 16) & 0xFFFFu) |
                  ((unsigned)NDIM << 16));             // dim1[31:16] | tile_dim0=256
    g1[4] = NTILE;                                     // tile_dim1=64 | tile_dim2=0
    g1[5] = NDIM;                                      // tensor_dim0_stride lo32
    g1[6] = 0;
    g1[7] = 0;

    v4i zz = {0, 0, 0, 0};
#if defined(__clang_major__) && __clang_major__ >= 23
    v8i z8 = {0, 0, 0, 0, 0, 0, 0, 0};
    __builtin_amdgcn_tensor_load_to_lds(g0, g1, zz, zz, z8, 0);
#else
    __builtin_amdgcn_tensor_load_to_lds(g0, g1, zz, zz, 0);
#endif
}
#else
// Fallback: cooperative synchronous copy of one 64x256 bf16 tile into LDS.
__device__ __forceinline__ void copy_stage(unsigned short* dst,
                                           const unsigned short* __restrict__ zn,
                                           int ct, int tid) {
    const int r = tid >> 2;                            // 0..63
    const int q = tid & 3;
    const unsigned short* src = zn + (size_t)(ct * NTILE + r) * NDIM + q * 64;
    unsigned short*       d   = dst + r * NDIM + q * 64;
#pragma unroll
    for (int u = 0; u < 8; ++u)
        *(uint4*)(d + u * 8) = *(const uint4*)(src + u * 8);
}
#endif

// ---------------------------------------------------------------------------
// Kernel 2: fused sim = Zn @ Zn^T (bf16 WMMA, f32 acc) + streaming softmax
// denominator (diag excluded) + target-logit capture, over this block's
// column range [csplit*2048, +2048). Double-buffered LDS B-tiles fed by the
// Tensor Data Mover; one barrier per 64-column tile. Logits bounded by
// 1/T = 2, so no running-max pass is needed. Partials go to fixed slots
// (no atomics -> deterministic).
// ---------------------------------------------------------------------------
__global__ __launch_bounds__(256) void ntxent_simloss(const unsigned short* __restrict__ zn,
                                                      float* __restrict__ Spart,
                                                      float* __restrict__ Tpart) {
    __shared__ unsigned short btile[2][NTILE * NDIM];  // 2 x 32 KB double buffer

    const int tid     = threadIdx.x;
    const int lane    = tid & 31;
    const int wave    = tid >> 5;
    const int half    = lane >> 4;
    const int l15     = lane & 15;
    const int rowBase = blockIdx.x * MTILE + wave * 16;
    const int split   = blockIdx.y;
    const int ct0     = split * (CTILES / CSPLIT);
    const int ct1     = ct0 + (CTILES / CSPLIT);

    // Compiler-visible stores into btile: without these, the TDM path has no
    // IR-level writes to the LDS array and LLVM undef-folds every ds_load.
    // The barrier orders these dummy stores before the first DMA write.
    btile[0][tid] = 0;
    btile[1][tid] = 0;
    __syncthreads();

    // A fragments: this wave's 16 rows x full K=256, resident in VGPRs.
    v16bf a[8];
    const unsigned short* arow = zn + (size_t)(rowBase + l15) * NDIM;
#pragma unroll
    for (int kk = 0; kk < 8; ++kk)
        a[kk] = *(const v16bf*)(arow + kk * 32 + half * 16);

    float accS[8], accT[8];
#pragma unroll
    for (int v = 0; v < 8; ++v) { accS[v] = 0.f; accT[v] = 0.f; }

    // Prologue: stage this split's first tile into buffer 0.
#if USE_TDM
    if (tid < 32) tdm_stage(zn, ct0, 0);
#else
    copy_stage(btile[0], zn, ct0, tid);
#endif

    for (int ct = ct0; ct < ct1; ++ct) {
        const int buf = (ct - ct0) & 1;
#if USE_TDM
        if (tid < 32) {
#if __has_builtin(__builtin_amdgcn_s_wait_tensorcnt)
            __builtin_amdgcn_s_wait_tensorcnt(0);
#else
            asm volatile("s_wait_tensorcnt 0x0" ::: "memory");
#endif
        }
#endif
        __syncthreads();
        asm volatile("" ::: "memory");   // DMA wrote LDS behind the compiler's back

        if (ct + 1 < ct1) {
#if USE_TDM
            if (tid < 32) tdm_stage(zn, ct + 1, buf ^ 1);
#else
            copy_stage(btile[buf ^ 1], zn, ct + 1, tid);
#endif
        }

#pragma unroll
        for (int sub = 0; sub < NTILE / 16; ++sub) {
            const int colBase = ct * NTILE + sub * 16;
            const unsigned short* brow = &btile[buf][(sub * 16 + l15) * NDIM];

            // Preload all 8 B fragments, then issue the 8 WMMAs.
            v16bf b[8];
#pragma unroll
            for (int kk = 0; kk < 8; ++kk)
                b[kk] = *(const v16bf*)(brow + kk * 32 + half * 16);

            v8f c = {};
#pragma unroll
            for (int kk = 0; kk < 8; ++kk)
                c = __builtin_amdgcn_wmma_f32_16x16x32_bf16(
                        false, a[kk], false, b[kk], (short)0, c, false, false);

            // C/D layout: lanes 0-15 -> M=v, N=lane; lanes 16-31 -> M=v+8, N=lane-16.
            const int nIdx = colBase + l15;
#pragma unroll
            for (int v = 0; v < 8; ++v) {
                const int mIdx = rowBase + half * 8 + v;
                float s = c[v] * INV_T;                // logit
                float e = __expf(s);                   // bounded by e^2
                if (mIdx != nIdx)        accS[v] += e; // diagonal masked (-inf)
                if (nIdx == (mIdx ^ 1))  accT[v] += s; // positive-pair logit
            }
        }
    }

    // Row partials live across the 16 lanes of each half; butterfly-reduce.
#pragma unroll
    for (int m = 8; m >= 1; m >>= 1) {
#pragma unroll
        for (int v = 0; v < 8; ++v) {
            accS[v] += __shfl_xor(accS[v], m, 32);
            accT[v] += __shfl_xor(accT[v], m, 32);
        }
    }

    float S = 0.f, T = 0.f;
#pragma unroll
    for (int v = 0; v < 8; ++v)
        if (l15 == v) { S = accS[v]; T = accT[v]; }

    if (l15 < 8) {
        const int row = rowBase + half * 8 + l15;
        Spart[split * NROWS + row] = S;
        Tpart[split * NROWS + row] = T;
    }
}

// ---------------------------------------------------------------------------
// Kernel 3: combine column-split partials per row, deterministic mean.
// ---------------------------------------------------------------------------
__global__ __launch_bounds__(256) void ntxent_finish(const float* __restrict__ Spart,
                                                     const float* __restrict__ Tpart,
                                                     float* __restrict__ out) {
    __shared__ float smem[256];
    float acc = 0.f;
    for (int row = threadIdx.x; row < NROWS; row += 256) {
        float S = 0.f, T = 0.f;
#pragma unroll
        for (int k = 0; k < CSPLIT; ++k) {
            S += Spart[k * NROWS + row];
            T += Tpart[k * NROWS + row];
        }
        acc += __logf(S) - T;                // per-row NT-Xent loss
    }
    smem[threadIdx.x] = acc;
    __syncthreads();
    for (int off = 128; off > 0; off >>= 1) {
        if (threadIdx.x < off) smem[threadIdx.x] += smem[threadIdx.x + off];
        __syncthreads();
    }
    if (threadIdx.x == 0) out[0] = smem[0] / (float)NROWS;
}

extern "C" void kernel_launch(void* const* d_in, const int* in_sizes, int n_in,
                              void* d_out, int out_size, void* d_ws, size_t ws_size,
                              hipStream_t stream) {
    (void)in_sizes; (void)n_in; (void)out_size; (void)ws_size;

    const float* z   = (const float*)d_in[0];
    float*       out = (float*)d_out;

    // Workspace layout: [0, 4MB) zn bf16; then S partials; then T partials.
    unsigned short* zn    = (unsigned short*)d_ws;
    float*          Spart = (float*)((char*)d_ws + (size_t)NROWS * NDIM * sizeof(unsigned short));
    float*          Tpart = Spart + (size_t)CSPLIT * NROWS;

    ntxent_normalize<<<NROWS / 8, 256, 0, stream>>>(z, zn);
    dim3 grid(NROWS / MTILE, CSPLIT);
    ntxent_simloss<<<grid, 256, 0, stream>>>(zn, Spart, Tpart);
    ntxent_finish<<<1, 256, 0, stream>>>(Spart, Tpart, out);
}
